// MambaModel_78314433675309
// MI455X (gfx1250) — compile-verified, matrix operand order checked by reading, not run on hardware
//
#include <hip/hip_runtime.h>
#include <hip/hip_bf16.h>
#include <math.h>

// ---------------- model dims ----------------
#define BN      2
#define LC      1024
#define DM      768
#define DS      16
#define DCONV   4
#define DI      1536          // 2*DM
#define DTR     48
#define DTPAD   64            // DTR padded to K multiple of 32
#define DOUT    1000
#define TOK     (BN*LC)       // 2048 tokens

typedef unsigned short u16;
typedef __attribute__((ext_vector_type(16))) __bf16 v16bf;
typedef __attribute__((ext_vector_type(8)))  float  v8f;

// round-to-nearest-even fp32 -> bf16 bits
__device__ __forceinline__ u16 f32_to_bf16(float f) {
  union { float f; unsigned u; } v; v.f = f;
  unsigned r = v.u + 0x7FFFu + ((v.u >> 16) & 1u);
  return (u16)(r >> 16);
}

// ---------------- fp32 -> bf16 convert w/ column slice + zero pad ----------
__global__ void cvt_bf16_pad(const float* __restrict__ src, u16* __restrict__ dst,
                             int rows, int sld, int coff, int kcopy, int kdst) {
  long long i = (long long)blockIdx.x * blockDim.x + threadIdx.x;
  long long total = (long long)rows * kdst;
  if (i >= total) return;
  int r = (int)(i / kdst), c = (int)(i % kdst);
  float v = (c < kcopy) ? src[(size_t)r * sld + coff + c] : 0.0f;
  dst[i] = f32_to_bf16(v);
}

// ---------------- WMMA GEMM: C(MxN) = A(MxK,bf16) * W(NxK,bf16)^T ----------
// K is a compile-time template parameter so the k-loop fully unrolls
// (16 v_wmma per chunk body) and B tiles live in distinct register sets.
// 4 waves/block; each wave owns 4 N-tiles; block shares one 16-row M-tile.
// A strip staged in LDS via global_load_async_to_lds_b128, double buffered.
#define GWAVES 4
#define NT 4
#define KC 128                 // max K-chunk staged per LDS buffer

union Tile { v16bf v; uint4 q[2]; };

template <int K>
__global__ void __launch_bounds__(GWAVES * 32)
wmma_gemm_bf16(const u16* __restrict__ A, const u16* __restrict__ W,
               float* __restrict__ C, int M, int N) {
  constexpr int KCE = (K < KC) ? K : KC;   // 128 or 64
  constexpr int NCH = K / KCE;             // 6 / 12 / 1
  constexpr int ALD = KCE + 8;             // padded LDS row stride (u16)

  __shared__ u16 As[2][16 * ALD];

  const int tid   = threadIdx.x;
  const int lane  = tid & 31;
  const int wave  = tid >> 5;
  const int ntiles = N >> 4;
  const int nt0   = (blockIdx.x * GWAVES + wave) * NT;
  const int mtile = blockIdx.y;
  const bool active = (nt0 < ntiles);      // wave-uniform

  const int k0 = (lane >> 4) << 3;         // 0 or 8 (16-bit A/B VGPR striping)

  // stage 16 x KCE A chunk (chunk index c) into As[buf] (async, ASYNCcnt)
  auto stage = [&](int buf, int c) {
    constexpr int UNITS = KCE * 2;         // # of 16B units in 16 x KCE
    constexpr int UPR   = KCE / 8;         // 16B units per row
    const u16* src = A + (size_t)(mtile << 4) * K + (size_t)c * KCE;
    #pragma unroll
    for (int u0 = 0; u0 < UNITS; u0 += GWAVES * 32) {
      int u = u0 + tid;
      int row = u / UPR, col = (u % UPR) << 3;
      unsigned lds = (unsigned)(size_t)(&As[buf][row * ALD + col]);
      const u16* g = src + (size_t)row * K + col;
      asm volatile("global_load_async_to_lds_b128 %0, %1, off"
                   :: "v"(lds), "v"(g) : "memory");
    }
  };

  v8f acc[NT];
  const u16* Wp[NT];
  #pragma unroll
  for (int t = 0; t < NT; ++t) {
    v8f z = {};
    acc[t] = z;
    int nc = nt0 + t; if (nc > ntiles - 1) nc = ntiles - 1;   // clamp OOB tiles
    Wp[t] = W + ((size_t)((nc << 4) + (lane & 15))) * K + k0;
  }

  stage(0, 0);

  int buf = 0;
  for (int c = 0; c < NCH; ++c) {
    // our chunk-c async loads are the only outstanding ones -> drain them
    asm volatile("s_wait_asynccnt 0x0" ::: "memory");
    __syncthreads();                    // all waves' chunk-c data in LDS;
                                        // all waves done reading buf^1
    if (c + 1 < NCH) stage(buf ^ 1, c + 1);   // overlap with compute below

    if (active) {
      const int kb = c * KCE;
      #pragma unroll
      for (int t = 0; t < NT; ++t)
        __builtin_prefetch(Wp[t] + kb + KCE, 0, 0);   // global_prefetch_b8

      #pragma unroll
      for (int kk = 0; kk < KCE; kk += 32) {          // constant trip: 4 (or 2)
        Tile a;
        const u16* ap = &As[buf][(lane & 15) * ALD + kk + k0];
        a.q[0] = *(const uint4*)ap;                   // ds_load_b128
        a.q[1] = *(const uint4*)(ap + 16);
        Tile bt[NT];
        #pragma unroll
        for (int t = 0; t < NT; ++t) {                // distinct regs per tile
          bt[t].q[0] = *(const uint4*)(Wp[t] + kb + kk);
          bt[t].q[1] = *(const uint4*)(Wp[t] + kb + kk + 16);
        }
        #pragma unroll
        for (int t = 0; t < NT; ++t)
          acc[t] = __builtin_amdgcn_wmma_f32_16x16x32_bf16(
              false, a.v, false, bt[t].v, (short)0, acc[t], false, false);
      }
    }
    buf ^= 1;
  }

  if (active) {
    // C layout: VGPR r -> row (lane<16 ? r : r+8), col = lane&15
    const int row0 = (mtile << 4) + ((lane >> 4) << 3);
    #pragma unroll
    for (int t = 0; t < NT; ++t) {
      if (nt0 + t < ntiles) {
        int col = ((nt0 + t) << 4) + (lane & 15);
        #pragma unroll
        for (int r = 0; r < 8; ++r)
          C[(size_t)(row0 + r) * N + col] = acc[t][r];
      }
    }
  }
}

// ---------------- depthwise causal conv (D_CONV=4) + bias + SiLU -----------
__global__ void conv_silu(const float* __restrict__ xz, const float* __restrict__ cw,
                          const float* __restrict__ cb, float* __restrict__ u) {
  long long i = (long long)blockIdx.x * blockDim.x + threadIdx.x;
  if (i >= (long long)TOK * DI) return;
  int d = (int)(i % DI);
  long long bl = i / DI;
  int l = (int)(bl % LC);
  long long b = bl / LC;
  const float* col = xz + (size_t)b * LC * (2 * DI) + d;
  float acc = cb[d];
  #pragma unroll
  for (int j = 0; j < DCONV; ++j) {
    int ls = l - (DCONV - 1) + j;
    if (ls >= 0) acc += cw[d * DCONV + j] * col[(size_t)ls * (2 * DI)];
  }
  u[i] = acc / (1.0f + __expf(-acc));         // SiLU
}

// ---------------- delta = softplus(x + bias) --------------------------------
__global__ void softplus_bias(float* __restrict__ delta, const float* __restrict__ bias) {
  long long i = (long long)blockIdx.x * blockDim.x + threadIdx.x;
  if (i >= (long long)TOK * DI) return;
  int d = (int)(i % DI);
  float v = delta[i] + bias[d];
  delta[i] = (v > 20.0f) ? v : log1pf(__expf(v));
}

// ---------------- selective scan --------------------------------------------
// one 16-lane group per (b,d); lane n holds h[n]; y via intra-group shfl_xor.
__global__ void selective_scan(const float* __restrict__ u, const float* __restrict__ delta,
                               const float* __restrict__ xdbl, const float* __restrict__ A_log,
                               const float* __restrict__ Dv, float* __restrict__ y) {
  int lane = threadIdx.x & 31;
  int grp = ((int)blockIdx.x * (blockDim.x >> 5) + (threadIdx.x >> 5)) * 2 + (lane >> 4);
  if (grp >= BN * DI) return;
  int d = grp % DI, b = grp / DI;
  int n = lane & 15;
  float A = -__expf(A_log[d * DS + n]);
  float Dd = Dv[d];
  float h = 0.0f;
  const float* xd = xdbl + (size_t)b * LC * (DTR + 2 * DS);
  size_t base = (size_t)b * LC * DI + d;
  for (int l = 0; l < LC; ++l) {
    float dl = delta[base + (size_t)l * DI];
    float ul = u[base + (size_t)l * DI];
    float bv = xd[l * (DTR + 2 * DS) + DTR + n];
    float cv = xd[l * (DTR + 2 * DS) + DTR + DS + n];
    h = __expf(dl * A) * h + (dl * ul) * bv;
    float yv = h * cv;
    yv += __shfl_xor(yv, 1);
    yv += __shfl_xor(yv, 2);
    yv += __shfl_xor(yv, 4);
    yv += __shfl_xor(yv, 8);
    if (n == 0) y[base + (size_t)l * DI] = yv + ul * Dd;
  }
}

// ---------------- y *= silu(res); convert to bf16 ---------------------------
__global__ void gate_to_bf16(const float* __restrict__ y, const float* __restrict__ xz,
                             u16* __restrict__ out) {
  long long i = (long long)blockIdx.x * blockDim.x + threadIdx.x;
  if (i >= (long long)TOK * DI) return;
  int d = (int)(i % DI);
  long long t = i / DI;
  float r = xz[(size_t)t * (2 * DI) + DI + d];
  float v = y[i] * (r / (1.0f + __expf(-r)));
  out[i] = f32_to_bf16(v);
}

// ---------------- layernorm on last token -----------------------------------
__global__ void ln_last(const float* __restrict__ x, const float* __restrict__ g,
                        const float* __restrict__ bta, float* __restrict__ out) {
  int b = blockIdx.x;
  const float* row = x + ((size_t)b * LC + (LC - 1)) * DM;
  __shared__ float red[16];
  float s = 0.0f, s2 = 0.0f;
  for (int i = threadIdx.x; i < DM; i += blockDim.x) { float v = row[i]; s += v; s2 += v * v; }
  for (int o = 16; o > 0; o >>= 1) { s += __shfl_xor(s, o); s2 += __shfl_xor(s2, o); }
  int w = threadIdx.x >> 5;
  if ((threadIdx.x & 31) == 0) { red[2 * w] = s; red[2 * w + 1] = s2; }
  __syncthreads();
  if (threadIdx.x == 0) {
    float ts = 0.0f, ts2 = 0.0f;
    for (int i = 0; i < (int)(blockDim.x >> 5); ++i) { ts += red[2 * i]; ts2 += red[2 * i + 1]; }
    float mu = ts / DM;
    red[0] = mu;
    red[1] = ts2 / DM - mu * mu;
  }
  __syncthreads();
  float mu = red[0], rstd = rsqrtf(red[1] + 1e-5f);
  for (int i = threadIdx.x; i < DM; i += blockDim.x)
    out[(size_t)b * DM + i] = (row[i] - mu) * rstd * g[i] + bta[i];
}

// ---------------- head: out[b][o] = ln[b].head_w[o] + head_b[o] ------------
__global__ void head_gemv(const float* __restrict__ ln, const float* __restrict__ Wh,
                          const float* __restrict__ bh, float* __restrict__ out) {
  int gid = (int)blockIdx.x * blockDim.x + threadIdx.x;
  int wid = gid >> 5, lane = gid & 31;
  if (wid >= BN * DOUT) return;
  int b = wid / DOUT, o = wid % DOUT;
  const float* xr = ln + (size_t)b * DM;
  const float* wr = Wh + (size_t)o * DM;
  float s = 0.0f;
  for (int k = lane; k < DM; k += 32) s += xr[k] * wr[k];
  for (int off = 16; off > 0; off >>= 1) s += __shfl_xor(s, off);
  if (lane == 0) out[(size_t)b * DOUT + o] = s + bh[o];
}

// ============================ host launcher =================================
struct LP {
  const float *ipw, *cw, *cb, *xpw, *dpw, *dpb, *alog, *dv, *opw;
};

static inline int ceil_div(long long a, int b) { return (int)((a + b - 1) / b); }

extern "C" void kernel_launch(void* const* d_in, const int* in_sizes, int n_in,
                              void* d_out, int out_size, void* d_ws, size_t ws_size,
                              hipStream_t stream) {
  (void)n_in; (void)out_size; (void)ws_size;
  auto din = [&](int i) { return (const float*)d_in[i]; };

  // ---- classify inputs (supports dict-insertion order and jax-tree order)
  LP lp[2];
  const float *x, *lng, *lnb, *hw, *hb;
  if (in_sizes[0] == BN * LC * DM) {
    // Scheme A: x, [in_proj_w,conv_w,conv_b,x_proj_w,dt_proj_w,dt_proj_b,A_log,D,out_proj_w]x2,
    //           ln_g, ln_b, head_w, head_b
    x = din(0);
    for (int l = 0; l < 2; ++l) {
      int b = 1 + 9 * l;
      lp[l].ipw = din(b + 0); lp[l].cw  = din(b + 1); lp[l].cb  = din(b + 2);
      lp[l].xpw = din(b + 3); lp[l].dpw = din(b + 4); lp[l].dpb = din(b + 5);
      lp[l].alog = din(b + 6); lp[l].dv = din(b + 7); lp[l].opw = din(b + 8);
    }
    lng = din(19); lnb = din(20); hw = din(21); hb = din(22);
  } else {
    // Scheme B (alphabetical tree flatten): head_b, head_w,
    //   [A_log,D,conv_b,conv_w,dt_proj_b,dt_proj_w,in_proj_w,out_proj_w,x_proj_w]x2,
    //   ln_b, ln_g, x
    hb = din(0); hw = din(1);
    for (int l = 0; l < 2; ++l) {
      int b = 2 + 9 * l;
      lp[l].alog = din(b + 0); lp[l].dv  = din(b + 1); lp[l].cb  = din(b + 2);
      lp[l].cw   = din(b + 3); lp[l].dpb = din(b + 4); lp[l].dpw = din(b + 5);
      lp[l].ipw  = din(b + 6); lp[l].opw = din(b + 7); lp[l].xpw = din(b + 8);
    }
    lnb = din(20); lng = din(21); x = din(22);
  }

  // ---- carve workspace (256B aligned)
  uintptr_t base = (uintptr_t)d_ws; size_t off = 0;
  auto alloc = [&](size_t bytes) -> void* {
    void* p = (void*)(base + off);
    off += (bytes + 255) & ~(size_t)255;
    return p;
  };
  float* x0    = (float*)alloc(sizeof(float) * TOK * DM);
  float* x1    = (float*)alloc(sizeof(float) * TOK * DM);
  float* xz    = (float*)alloc(sizeof(float) * TOK * 2 * DI);
  float* ubuf  = (float*)alloc(sizeof(float) * TOK * DI);
  float* xdbl  = (float*)alloc(sizeof(float) * TOK * (DTR + 2 * DS));
  float* delta = (float*)alloc(sizeof(float) * TOK * DI);
  float* ysc   = (float*)alloc(sizeof(float) * TOK * DI);
  u16*   abf   = (u16*)alloc(sizeof(u16) * TOK * DI);            // activation bf16
  u16*   wbf   = (u16*)alloc(sizeof(u16) * (size_t)2 * DI * DM); // weight bf16 (max)
  float* lnbuf = (float*)alloc(sizeof(float) * BN * DM);

  const int EB = 256;
  auto cvt = [&](const float* s, u16* d, int rows, int sld, int coff, int kc, int kd) {
    cvt_bf16_pad<<<ceil_div((long long)rows * kd, EB), EB, 0, stream>>>(s, d, rows, sld, coff, kc, kd);
  };
  auto gemm = [&](const u16* A, const u16* W, float* C, int M, int N, int K) {
    dim3 grid(ceil_div(N >> 4, GWAVES * NT), M >> 4);
    if (K == DM)
      wmma_gemm_bf16<DM><<<grid, GWAVES * 32, 0, stream>>>(A, W, C, M, N);
    else if (K == DI)
      wmma_gemm_bf16<DI><<<grid, GWAVES * 32, 0, stream>>>(A, W, C, M, N);
    else
      wmma_gemm_bf16<DTPAD><<<grid, GWAVES * 32, 0, stream>>>(A, W, C, M, N);
  };

  const float* xin = x;
  float* xouts[2] = { x0, x1 };

  for (int l = 0; l < 2; ++l) {
    const LP& P = lp[l];
    // in_proj: xz = xin @ in_proj_w^T       (2048 x 3072 x 768)
    cvt(xin,   abf, TOK,     DM, 0, DM, DM);
    cvt(P.ipw, wbf, 2 * DI,  DM, 0, DM, DM);
    gemm(abf, wbf, xz, TOK, 2 * DI, DM);
    // depthwise conv + SiLU -> u
    conv_silu<<<ceil_div((long long)TOK * DI, EB), EB, 0, stream>>>(xz, P.cw, P.cb, ubuf);
    // x_proj: xdbl = u @ x_proj_w^T         (2048 x 80 x 1536)
    cvt(ubuf,  abf, TOK, DI, 0, DI, DI);
    cvt(P.xpw, wbf, DTR + 2 * DS, DI, 0, DI, DI);
    gemm(abf, wbf, xdbl, TOK, DTR + 2 * DS, DI);
    // dt_proj: delta = dt @ dt_proj_w^T     (2048 x 1536 x 48->64 padded)
    cvt(xdbl,  abf, TOK, DTR + 2 * DS, 0, DTR, DTPAD);
    cvt(P.dpw, wbf, DI,  DTR, 0, DTR, DTPAD);
    gemm(abf, wbf, delta, TOK, DI, DTPAD);
    softplus_bias<<<ceil_div((long long)TOK * DI, EB), EB, 0, stream>>>(delta, P.dpb);
    // selective scan
    selective_scan<<<ceil_div((long long)BN * DI * 16, EB), EB, 0, stream>>>(
        ubuf, delta, xdbl, P.alog, P.dv, ysc);
    // gate + convert
    gate_to_bf16<<<ceil_div((long long)TOK * DI, EB), EB, 0, stream>>>(ysc, xz, abf);
    // out_proj: xout = y @ out_proj_w^T     (2048 x 768 x 1536)
    cvt(P.opw, wbf, DM, DI, 0, DI, DI);
    gemm(abf, wbf, xouts[l], TOK, DM, DI);
    xin = xouts[l];
  }

  // LN(last token) + head
  ln_last<<<BN, 256, 0, stream>>>(xin, lng, lnb, lnbuf);
  head_gemv<<<ceil_div((long long)BN * DOUT * 32, EB), EB, 0, stream>>>(
      lnbuf, hw, hb, (float*)d_out);
}
